// DynamicGNN_26697516712152
// MI455X (gfx1250) — matrix operand order repaired and checked by reading, not compiled, not run on hardware
//
#include <hip/hip_runtime.h>
#include <hip/hip_bf16.h>
#include <cfloat>

// ---------------------------------------------------------------------------
// DynamicGNN (TransformerConv x3) for MI455X / gfx1250, wave32 + WMMA bf16.
// GEMMs: v_wmma_f32_16x16x32_bf16, weights pre-swizzled to fragment-major so
// B-frags are 2x ds_load_b128 and A-frags are 2x global_load_b128 per lane.
// Epilogue: stripe-level bounds test + immediate-offset stores.
// Softmax / segment reductions kept fp32 with atomics (HBM/scatter bound).
// ---------------------------------------------------------------------------

typedef __bf16 bf16_t;
typedef bf16_t v16bf __attribute__((ext_vector_type(16)));
typedef float  v8f   __attribute__((ext_vector_type(8)));

union Frag2 { v16bf v; uint4 q[2]; };

static __device__ __forceinline__ unsigned short f2bf(float f) {
    unsigned u = __float_as_uint(f);
    unsigned r = u + 0x7FFFu + ((u >> 16) & 1u);   // round-to-nearest-even
    return (unsigned short)(r >> 16);
}

// monotonic encoding of float into uint so atomicMax(uint) == float max
static __device__ __forceinline__ unsigned enc_f(float f) {
    unsigned u = __float_as_uint(f);
    return (u & 0x80000000u) ? ~u : (u | 0x80000000u);
}
static __device__ __forceinline__ float dec_f(unsigned u) {
    u = (u & 0x80000000u) ? (u & 0x7FFFFFFFu) : ~u;
    return __uint_as_float(u);
}

// ---------------------------------------------------------------------------
// Weight convert fp32 -> bf16 with fragment-major swizzle.
// B (32x16 bf16) lane layout: lane<16: col = lane,    K = kb*32 + 0..15
//                             lane>=16: col = lane-16, K = kb*32 + 16..31
// Swizzled index: Wsw[((nt*4+kb)*32 + lane)*16 + i]
// ---------------------------------------------------------------------------
__global__ void k_swz128(const float* __restrict__ W,        // [L][128][128] (K x N)
                         unsigned short* __restrict__ Wsw,   // same element count
                         long long total) {
    long long t = (long long)blockIdx.x * blockDim.x + threadIdx.x;
    if (t >= total) return;
    long long l = t >> 14;            // layer
    int r = (int)(t & 16383);
    int frag = r >> 9;                // 0..31  (= nt*4 + kb)
    int lane = (r >> 4) & 31;
    int i    = r & 15;
    int nt = frag >> 2;
    int kb = frag & 3;
    int col = nt * 16 + (lane & 15);
    int k   = kb * 32 + ((lane >> 4) & 1) * 16 + i;
    Wsw[t] = f2bf(W[l * 16384 + k * 128 + col]);
}

__global__ void k_swz_out(const float* __restrict__ W,       // [128][16]
                          unsigned short* __restrict__ Wsw) { // 2048 elems
    int t = blockIdx.x * blockDim.x + threadIdx.x;
    if (t >= 2048) return;
    int kb   = t >> 9;
    int lane = (t >> 4) & 31;
    int i    = t & 15;
    int col = lane & 15;
    int k   = kb * 32 + ((lane >> 4) & 1) * 16 + i;
    Wsw[t] = f2bf(W[k * 16 + col]);
}

// ---------------------------------------------------------------------------
// Feature build: concat(attr[96], sinusoidal PE of two timestamps[2x16]) -> bf16
// pe[t][2i] = sin(t*div_i), pe[t][2i+1] = cos(t*div_i), div_i = exp(-ln(1e4)*2i/16)
// ---------------------------------------------------------------------------
static __device__ __forceinline__ float pe_val(float ts, int j) {
    int i = j >> 1;
    float freq = __expf(-logf(10000.0f) * (2.0f * (float)i) / 16.0f);
    float ang = ts * freq;
    return (j & 1) ? __cosf(ang) : __sinf(ang);
}

__global__ void k_feat(const float* __restrict__ attr, const int* __restrict__ lu,
                       unsigned short* __restrict__ out, long long total) {
    long long t = (long long)blockIdx.x * blockDim.x + threadIdx.x;
    if (t >= total) return;
    long long n = t >> 7;
    int c = (int)(t & 127);
    float v;
    if (c < 96) {
        v = attr[n * 96 + c];
    } else {
        int idx = c - 96;
        int which = idx >> 4;      // 0 or 1
        int j = idx & 15;
        float ts = (float)lu[n * 2 + which];
        v = pe_val(ts, j);
    }
    out[t] = f2bf(v);
}

// ---------------------------------------------------------------------------
// Fragment loaders.
// A (16x32 bf16) loaded layout: lane<16: rows 0..15, K in kb+0..15
//                            lane>=16: same rows,  K in kb+16..31
// -> one contiguous 32B run per lane (two adjacent 16B chunks): 2x global_load_b128.
// ---------------------------------------------------------------------------
static __device__ __forceinline__ v16bf load_a_frag(const unsigned short* __restrict__ A,
                                                    int row_base, int M, int kb,
                                                    unsigned lane) {
    int row = row_base + (int)(lane & 15u);
    if (row >= M) row = M - 1;                       // clamp (output is guarded)
    int kbase = kb * 32 + (int)((lane >> 4) & 1u) * 16;
    const uint4* p = (const uint4*)(A + (long long)row * 128 + kbase);
    Frag2 f;
    f.q[0] = p[0];     // K = kbase + 0..7
    f.q[1] = p[1];     // K = kbase + 8..15
    return f.v;
}

// B fragment from fragment-major LDS: 32 contiguous bytes per lane.
static __device__ __forceinline__ v16bf load_b_frag(const unsigned short* __restrict__ lWs,
                                                    int frag, unsigned lane) {
    const uint4* p = (const uint4*)lWs + ((frag * 32 + (int)lane) << 1);
    Frag2 f;
    f.q[0] = p[0];
    f.q[1] = p[1];
    return f.v;
}

// ---------------------------------------------------------------------------
// C[M,128] = A[M,128](bf16) @ W[128,128](bf16, fragment-major) + bias
// 256 threads = 8 waves; each wave owns a 16-row stripe; W staged in LDS (32KB)
// ---------------------------------------------------------------------------
__global__ __launch_bounds__(256) void k_gemm128(const unsigned short* __restrict__ A,
                                                 const unsigned short* __restrict__ Wsw,
                                                 const float* __restrict__ bias,
                                                 float* __restrict__ C, int M) {
    __shared__ unsigned short lW[128 * 128];
    int tid = threadIdx.x;
    { // cooperative copy 32KB of swizzled weights (2048 uint4)
        const uint4* src = (const uint4*)Wsw;
        uint4* dst = (uint4*)lW;
        for (int i = tid; i < 2048; i += 256) dst[i] = src[i];
    }
    __syncthreads();

    int wave = tid >> 5;
    unsigned lane = (unsigned)(tid & 31);
    int row_base = blockIdx.x * 128 + wave * 16;
    if (row_base >= M) return;

    v16bf a[4];
#pragma unroll
    for (int kb = 0; kb < 4; kb++) a[kb] = load_a_frag(A, row_base, M, kb, lane);

    // all 8 n-tile accumulators first; stores happen once at the end
    v8f acc[8];
#pragma unroll
    for (int nt = 0; nt < 8; nt++) {
        acc[nt] = (v8f){};
#pragma unroll
        for (int kb = 0; kb < 4; kb++) {
            v16bf b = load_b_frag(lW, nt * 4 + kb, lane);
            acc[nt] = __builtin_amdgcn_wmma_f32_16x16x32_bf16(
                false, a[kb], false, b, (short)0, acc[nt], false, false);
        }
    }

    int rofs = (int)((lane >> 4) & 1u) * 8;
    int cofs = (int)(lane & 15u);

    // hoisted bias (one branch per wave)
    float bb[8];
    if (bias) {
#pragma unroll
        for (int nt = 0; nt < 8; nt++) bb[nt] = bias[nt * 16 + cofs];
    } else {
#pragma unroll
        for (int nt = 0; nt < 8; nt++) bb[nt] = 0.0f;
    }

    float* base = C + (long long)(row_base + rofs) * 128 + cofs;
    if (row_base + 16 <= M) {
        // full stripe: unguarded stores, all immediate offsets from one base
#pragma unroll
        for (int nt = 0; nt < 8; nt++)
#pragma unroll
            for (int r = 0; r < 8; r++)
                base[r * 128 + nt * 16] = acc[nt][r] + bb[nt];
    } else {
        // tail stripe (at most one workgroup)
#pragma unroll
        for (int nt = 0; nt < 8; nt++)
#pragma unroll
            for (int r = 0; r < 8; r++)
                if (row_base + rofs + r < M)
                    base[r * 128 + nt * 16] = acc[nt][r] + bb[nt];
    }
}

// ---------------------------------------------------------------------------
// out[M,16] = H[M,128](bf16) @ Wo[128,16](bf16, fragment-major) + bout
// ---------------------------------------------------------------------------
__global__ __launch_bounds__(256) void k_out_gemm(const unsigned short* __restrict__ H,
                                                  const unsigned short* __restrict__ Wosw,
                                                  const float* __restrict__ bout,
                                                  float* __restrict__ out, int M) {
    __shared__ unsigned short lW[2048];
    int tid = threadIdx.x;
    {
        const uint4* src = (const uint4*)Wosw;
        uint4* dst = (uint4*)lW;
        for (int i = tid; i < 256; i += 256) dst[i] = src[i];
    }
    __syncthreads();

    int wave = tid >> 5;
    unsigned lane = (unsigned)(tid & 31);
    int row_base = blockIdx.x * 128 + wave * 16;
    if (row_base >= M) return;

    v8f acc = {};
#pragma unroll
    for (int kb = 0; kb < 4; kb++) {
        v16bf a = load_a_frag(H, row_base, M, kb, lane);
        v16bf b = load_b_frag(lW, kb, lane);
        acc = __builtin_amdgcn_wmma_f32_16x16x32_bf16(
            false, a, false, b, (short)0, acc, false, false);
    }
    int cofs = (int)(lane & 15u);
    int rofs = (int)((lane >> 4) & 1u) * 8;
    float bb = bout[cofs];
    float* base = out + (long long)(row_base + rofs) * 16 + cofs;
    if (row_base + 16 <= M) {
#pragma unroll
        for (int r = 0; r < 8; r++) base[r * 16] = acc[r] + bb;
    } else {
#pragma unroll
        for (int r = 0; r < 8; r++)
            if (row_base + rofs + r < M) base[r * 16] = acc[r] + bb;
    }
}

// ---------------------------------------------------------------------------
// Per-block scratch init: m = enc(-FLT_MAX), denom = 0, agg = 0
// ---------------------------------------------------------------------------
__global__ void k_init(unsigned* __restrict__ m, float* __restrict__ denom,
                       float* __restrict__ agg, long long n128, long long n8) {
    long long t = (long long)blockIdx.x * blockDim.x + threadIdx.x;
    if (t < n8) { m[t] = enc_f(-FLT_MAX); denom[t] = 0.0f; }
    if (t < n128) agg[t] = 0.0f;
}

// ---------------------------------------------------------------------------
// Edge pass 1: alpha[e,h] = scale * q[dst]·(k[src]+e); segment max via atomicMax
// one thread per (edge, head)
// ---------------------------------------------------------------------------
__global__ void k_alpha(const float* __restrict__ q, const float* __restrict__ k,
                        const float* __restrict__ efeat,
                        const int* __restrict__ src, const int* __restrict__ dst,
                        float* __restrict__ alpha, unsigned* __restrict__ m_enc,
                        long long total) {
    long long t = (long long)blockIdx.x * blockDim.x + threadIdx.x;
    if (t >= total) return;
    long long eid = t >> 3;
    int h = (int)(t & 7);
    int s = src[eid], d = dst[eid];
    const float* qp = q + (long long)d * 128 + h * 16;
    const float* kp = k + (long long)s * 128 + h * 16;
    const float* ep = efeat + eid * 128 + h * 16;
    float acc = 0.0f;
#pragma unroll
    for (int c = 0; c < 16; c++) acc += qp[c] * (kp[c] + ep[c]);
    acc *= 0.25f;                                    // 1/sqrt(16)
    alpha[eid * 8 + h] = acc;
    atomicMax(&m_enc[(long long)d * 8 + h], enc_f(acc));
}

// ---------------------------------------------------------------------------
// Edge pass 2: p = exp(alpha - m); denom += p; agg += p * (v[src]+e)
// ---------------------------------------------------------------------------
__global__ void k_scatter(const float* __restrict__ alpha, const unsigned* __restrict__ m_enc,
                          const float* __restrict__ v, const float* __restrict__ efeat,
                          const int* __restrict__ src, const int* __restrict__ dst,
                          float* __restrict__ denom, float* __restrict__ agg,
                          long long total) {
    long long t = (long long)blockIdx.x * blockDim.x + threadIdx.x;
    if (t >= total) return;
    long long eid = t >> 3;
    int h = (int)(t & 7);
    int s = src[eid], d = dst[eid];
    float m = dec_f(m_enc[(long long)d * 8 + h]);
    float p = __expf(alpha[eid * 8 + h] - m);
    atomicAdd(&denom[(long long)d * 8 + h], p);
    const float* vp = v + (long long)s * 128 + h * 16;
    const float* ep = efeat + eid * 128 + h * 16;
    float* ap = agg + (long long)d * 128 + h * 16;
#pragma unroll
    for (int c = 0; c < 16; c++) atomicAdd(&ap[c], p * (vp[c] + ep[c]));
}

// ---------------------------------------------------------------------------
// Node update: h = relu(agg/(denom+1e-16) + skip), re-emit bf16
// ---------------------------------------------------------------------------
__global__ void k_node_update(const float* __restrict__ agg, const float* __restrict__ denom,
                              const float* __restrict__ skip,
                              unsigned short* __restrict__ hbf, long long n128) {
    long long t = (long long)blockIdx.x * blockDim.x + threadIdx.x;
    if (t >= n128) return;
    long long node = t >> 7;
    int c = (int)(t & 127);
    float d = denom[node * 8 + (c >> 4)];
    float val = agg[t] / (d + 1e-16f) + skip[t];
    val = val > 0.0f ? val : 0.0f;
    hbf[t] = f2bf(val);
}

// ---------------------------------------------------------------------------
// Host orchestration
// ---------------------------------------------------------------------------
static inline long long cdiv(long long a, long long b) { return (a + b - 1) / b; }

extern "C" void kernel_launch(void* const* d_in, const int* in_sizes, int n_in,
                              void* d_out, int out_size, void* d_ws, size_t ws_size,
                              hipStream_t stream) {
    const float* x      = (const float*)d_in[0];
    const int*   nlu    = (const int*)d_in[1];
    const int*   eidx   = (const int*)d_in[2];
    const float* eattr  = (const float*)d_in[3];
    const int*   elu    = (const int*)d_in[4];
    const float* Wq     = (const float*)d_in[5];
    const float* bq     = (const float*)d_in[6];
    const float* Wk     = (const float*)d_in[7];
    const float* bk     = (const float*)d_in[8];
    const float* Wv     = (const float*)d_in[9];
    const float* bv     = (const float*)d_in[10];
    const float* We     = (const float*)d_in[11];
    const float* Ws     = (const float*)d_in[12];
    const float* bs     = (const float*)d_in[13];
    const float* Wout   = (const float*)d_in[14];
    const float* bout   = (const float*)d_in[15];

    const long long N = in_sizes[0] / 96;   // 100000
    const long long E = in_sizes[3] / 96;   // 800000
    const int* src = eidx;
    const int* dst = eidx + E;

    // workspace carve-up (256B aligned)
    char* p = (char*)d_ws;
    auto carve = [&](size_t bytes) -> void* {
        void* r = (void*)p;
        p += (bytes + 255) & ~(size_t)255;
        return r;
    };
    unsigned short* Wq_b = (unsigned short*)carve(3 * 16384 * 2);
    unsigned short* Wk_b = (unsigned short*)carve(3 * 16384 * 2);
    unsigned short* Wv_b = (unsigned short*)carve(3 * 16384 * 2);
    unsigned short* We_b = (unsigned short*)carve(3 * 16384 * 2);
    unsigned short* Ws_b = (unsigned short*)carve(3 * 16384 * 2);
    unsigned short* Wo_b = (unsigned short*)carve(2048 * 2);
    unsigned short* h_b  = (unsigned short*)carve((size_t)N * 128 * 2);
    unsigned short* ea_b = (unsigned short*)carve((size_t)E * 128 * 2);
    float* qf    = (float*)carve((size_t)N * 128 * 4);
    float* kf    = (float*)carve((size_t)N * 128 * 4);
    float* vf    = (float*)carve((size_t)N * 128 * 4);
    float* sf    = (float*)carve((size_t)N * 128 * 4);
    float* ef    = (float*)carve((size_t)E * 128 * 4);
    float* alpha = (float*)carve((size_t)E * 8 * 4);
    unsigned* m  = (unsigned*)carve((size_t)N * 8 * 4);
    float* denom = (float*)carve((size_t)N * 8 * 4);
    float* agg   = (float*)carve((size_t)N * 128 * 4);
    (void)ws_size;

    const int T = 256;

    // weights -> bf16, fragment-major swizzle
    k_swz128<<<cdiv(3 * 16384, T), T, 0, stream>>>(Wq, Wq_b, 3 * 16384);
    k_swz128<<<cdiv(3 * 16384, T), T, 0, stream>>>(Wk, Wk_b, 3 * 16384);
    k_swz128<<<cdiv(3 * 16384, T), T, 0, stream>>>(Wv, Wv_b, 3 * 16384);
    k_swz128<<<cdiv(3 * 16384, T), T, 0, stream>>>(We, We_b, 3 * 16384);
    k_swz128<<<cdiv(3 * 16384, T), T, 0, stream>>>(Ws, Ws_b, 3 * 16384);
    k_swz_out<<<cdiv(2048, T), T, 0, stream>>>(Wout, Wo_b);

    // features (attr + positional encoding) -> bf16
    k_feat<<<cdiv(N * 128, T), T, 0, stream>>>(x, nlu, h_b, N * 128);
    k_feat<<<cdiv(E * 128, T), T, 0, stream>>>(eattr, elu, ea_b, E * 128);

    const int gN = (int)cdiv(N, 128);
    const int gE = (int)cdiv(E, 128);

    for (int l = 0; l < 3; l++) {
        const unsigned short* wq = Wq_b + (size_t)l * 16384;
        const unsigned short* wk = Wk_b + (size_t)l * 16384;
        const unsigned short* wv = Wv_b + (size_t)l * 16384;
        const unsigned short* we = We_b + (size_t)l * 16384;
        const unsigned short* ws = Ws_b + (size_t)l * 16384;

        k_gemm128<<<gN, T, 0, stream>>>(h_b, wq, bq + l * 128, qf, (int)N);
        k_gemm128<<<gN, T, 0, stream>>>(h_b, wk, bk + l * 128, kf, (int)N);
        k_gemm128<<<gN, T, 0, stream>>>(h_b, wv, bv + l * 128, vf, (int)N);
        k_gemm128<<<gN, T, 0, stream>>>(h_b, ws, bs + l * 128, sf, (int)N);
        k_gemm128<<<gE, T, 0, stream>>>(ea_b, we, nullptr, ef, (int)E);

        k_init<<<cdiv(N * 128, T), T, 0, stream>>>(m, denom, agg, N * 128, N * 8);
        k_alpha<<<cdiv(E * 8, T), T, 0, stream>>>(qf, kf, ef, src, dst, alpha, m, E * 8);
        k_scatter<<<cdiv(E * 8, T), T, 0, stream>>>(alpha, m, vf, ef, src, dst, denom, agg, E * 8);
        k_node_update<<<cdiv(N * 128, T), T, 0, stream>>>(agg, denom, sf, h_b, N * 128);
    }

    k_out_gemm<<<gN, T, 0, stream>>>(h_b, Wo_b, bout, (float*)d_out, (int)N);
}
